// CustomLSTMModel_1108101563124
// MI455X (gfx1250) — matrix operand order repaired and checked by reading, not compiled, not run on hardware
//
#include <hip/hip_runtime.h>
#include <hip/hip_bf16.h>
#include <math.h>

// ---------------------------------------------------------------------------
// Problem constants (from reference)
// ---------------------------------------------------------------------------
#define BATCH   64
#define SEQLEN  512
#define EMBED   512
#define HIDDEN  1024
#define VOCAB   32000

typedef __attribute__((ext_vector_type(16))) _Float16 v16h;
typedef __attribute__((ext_vector_type(8)))  float    v8f;

union FragU {
    v16h h;
    uint4 q[2];
};

__device__ __forceinline__ uint4 ld16B(const _Float16* p) {
    return *reinterpret_cast<const uint4*>(p);
}

__device__ __forceinline__ float sigmoidf_(float x) {
    return 1.0f / (1.0f + expf(-x));
}

// ---------------------------------------------------------------------------
// fp32 -> fp16 conversion (weights), grid-stride
// ---------------------------------------------------------------------------
__global__ void convert_f32_f16_kernel(const float* __restrict__ in,
                                       _Float16* __restrict__ out, size_t n) {
    size_t i = (size_t)blockIdx.x * blockDim.x + threadIdx.x;
    size_t stride = (size_t)gridDim.x * blockDim.x;
    for (; i < n; i += stride) out[i] = (_Float16)in[i];
}

// ---------------------------------------------------------------------------
// Embedding gather + fp16 convert: xs[t][b][e] = emb[texts[b][t]][e]
// one block per (t,b)
// ---------------------------------------------------------------------------
__global__ __launch_bounds__(128)
void embed_kernel(const int* __restrict__ texts, const float* __restrict__ emb,
                  _Float16* __restrict__ xs) {
    int tb = blockIdx.x;                 // t*BATCH + b
    int t = tb / BATCH, b = tb % BATCH;
    int tok = texts[b * SEQLEN + t];
    const float* src = emb + (size_t)tok * EMBED;
    _Float16* dst = xs + (size_t)tb * EMBED;
    for (int e = threadIdx.x; e < EMBED; e += blockDim.x)
        dst[e] = (_Float16)src[e];
}

// ---------------------------------------------------------------------------
// GEMM span helper: accumulate acc[g*2+m] += A_m(16x32) * B_g(32x16) over
// k0 in [kBeg, kEnd).  abase0/abase1 are pre-biased so (abase + k0 + aK) is
// the correct element address for the whole span (no per-iteration select).
// ---------------------------------------------------------------------------
__device__ __forceinline__
void gemm_span(const _Float16* __restrict__ abase0,
               const _Float16* __restrict__ abase1,
               int kBeg, int kEnd, int aK, int bK,
               const _Float16* __restrict__ wb0, const _Float16* __restrict__ wb1,
               const _Float16* __restrict__ wb2, const _Float16* __restrict__ wb3,
               v8f acc[8])
{
    for (int k0 = kBeg; k0 < kEnd; k0 += 32) {
        FragU a0, a1;
        const _Float16* ap0 = abase0 + k0 + aK;
        const _Float16* ap1 = abase1 + k0 + aK;
        a0.q[0] = ld16B(ap0); a0.q[1] = ld16B(ap0 + 16);
        a1.q[0] = ld16B(ap1); a1.q[1] = ld16B(ap1 + 16);

        const _Float16* wb[4] = { wb0, wb1, wb2, wb3 };
#pragma unroll
        for (int g = 0; g < 4; ++g) {
            FragU b;
            const _Float16* wp = wb[g] + k0 + bK;
            b.q[0] = ld16B(wp); b.q[1] = ld16B(wp + 16);
            acc[g * 2 + 0] = __builtin_amdgcn_wmma_f32_16x16x32_f16(
                false, a0.h, false, b.h, (short)0, acc[g * 2 + 0], false, false);
            acc[g * 2 + 1] = __builtin_amdgcn_wmma_f32_16x16x32_f16(
                false, a1.h, false, b.h, (short)0, acc[g * 2 + 1], false, false);
        }
    }
}

// ---------------------------------------------------------------------------
// Fused LSTM cell step:  gate = sigma/tanh( [h|x] @ W^T + b ), c/h update.
// One wave (32 threads) per 32x16 (batch, hidden) macro-tile: 2 batch tiles
// x 4 gates = 8 register accumulator tiles, so each gate-weight fragment is
// fetched once per 2 batch tiles (halved L2 weight traffic), and the whole
// cell nonlinearity is applied straight out of the WMMA accumulators.
// Grid: (BATCH/32, HIDDEN/16) = (2, 64), block: 32 (one wave32).
// ---------------------------------------------------------------------------
__global__ __launch_bounds__(32)
void lstm_step_kernel(const _Float16* __restrict__ x, int F,   // (B, F) fp16
                      const _Float16* __restrict__ hprev,      // (B, H) fp16
                      const float* cprev,                      // (B, H) f32 (in-place ok)
                      const _Float16* __restrict__ Wf,
                      const _Float16* __restrict__ Wi,
                      const _Float16* __restrict__ Wo,
                      const _Float16* __restrict__ Wc,         // (H, H+F) fp16, cols [h|x]
                      const float* __restrict__ bf, const float* __restrict__ bi,
                      const float* __restrict__ bo, const float* __restrict__ bc,
                      _Float16* __restrict__ hout,             // (B, H) fp16
                      float* cout)                             // (B, H) f32
{
    const int H = HIDDEN;
    const int K = H + F;

    const int lane  = threadIdx.x;       // 0..31
    const int mg    = blockIdx.x;        // 0..1   (pair of batch tiles)
    const int htile = blockIdx.y;        // 0..63  (hidden tiles)

    // A fragment addressing (16x32 f16): lane l holds row (l&15),
    // K chunks [aK, aK+8) and [aK+16, aK+24) with aK = (l>=16)*8.
    const int aRow0 = mg * 32 + (lane & 15);
    const int aRow1 = aRow0 + 16;
    const int aK    = (lane >> 4) * 8;
    // B fragment addressing (32x16 f16): lane l holds column (l&15),
    // K chunks [bK, bK+8) and [bK+16, bK+24) with bK = (l>=16)*8.
    const int wRow = htile * 16 + (lane & 15);
    const int bK   = (lane >> 4) * 8;

    // Pre-biased A bases: h-span indexes with k0 in [0,H), x-span with
    // k0 in [H,K) (so subtract H from the x base once).
    const _Float16* hb0 = hprev + (size_t)aRow0 * H;
    const _Float16* hb1 = hprev + (size_t)aRow1 * H;
    const _Float16* xb0 = x + (size_t)aRow0 * F - H;
    const _Float16* xb1 = x + (size_t)aRow1 * F - H;

    const size_t wOff = (size_t)wRow * K;
    const _Float16* wbF = Wf + wOff;
    const _Float16* wbI = Wi + wOff;
    const _Float16* wbO = Wo + wOff;
    const _Float16* wbC = Wc + wOff;

    v8f acc[8];
#pragma unroll
    for (int i = 0; i < 8; ++i) acc[i] = (v8f){};

    // [h | x] concat handled as two clean spans (no per-iteration select).
    gemm_span(hb0, hb1, 0, H, aK, bK, wbF, wbI, wbO, wbC, acc);
    gemm_span(xb0, xb1, H, K, aK, bK, wbF, wbI, wbO, wbC, acc);

    // ---- fused cell update from WMMA D layout:
    // VGPR v of lane l holds element (M = v + 8*(l>=16), N = l&15) of a tile.
    const int col = htile * 16 + (lane & 15);
    const float bF = bf[col], bI = bi[col], bO = bo[col], bC = bc[col];

#pragma unroll
    for (int m = 0; m < 2; ++m) {
        const int rbase = mg * 32 + m * 16 + (lane >> 4) * 8;
#pragma unroll
        for (int v = 0; v < 8; ++v) {
            const size_t idx = (size_t)(rbase + v) * H + col;
            float ft = sigmoidf_(acc[0 * 2 + m][v] + bF);
            float it = sigmoidf_(acc[1 * 2 + m][v] + bI);
            float ot = sigmoidf_(acc[2 * 2 + m][v] + bO);
            float ch = tanhf(acc[3 * 2 + m][v] + bC);
            float c  = ft * cprev[idx] + it * ch;
            cout[idx] = c;
            hout[idx] = (_Float16)(ot * tanhf(c));
        }
    }
}

// ---------------------------------------------------------------------------
// Final projection: y = h_last @ Wy^T + by  (64 x 32000), K = 1024.
// 2 batch tiles per wave (Wy fragment fetched once per 2 batch tiles).
// Grid: (2, 2000), block: 32.
// ---------------------------------------------------------------------------
__global__ __launch_bounds__(32)
void proj_kernel(const _Float16* __restrict__ h,    // (B, H) fp16
                 const _Float16* __restrict__ Wy,   // (V, H) fp16
                 const float* __restrict__ by,
                 float* __restrict__ out)           // (B, V) f32
{
    const int K = HIDDEN;
    const int lane  = threadIdx.x;
    const int mg    = blockIdx.x;     // 0..1
    const int ntile = blockIdx.y;     // 0..1999

    const int aRow0 = mg * 32 + (lane & 15);
    const int aRow1 = aRow0 + 16;
    const int aK    = (lane >> 4) * 8;
    const int wRow  = ntile * 16 + (lane & 15);
    const int bK    = (lane >> 4) * 8;

    const _Float16* ab0 = h + (size_t)aRow0 * K;
    const _Float16* ab1 = h + (size_t)aRow1 * K;
    const _Float16* wrow = Wy + (size_t)wRow * K;

    v8f acc0 = {}, acc1 = {};
    for (int k0 = 0; k0 < K; k0 += 32) {
        FragU a0, a1, b;
        const _Float16* ap0 = ab0 + k0 + aK;
        const _Float16* ap1 = ab1 + k0 + aK;
        a0.q[0] = ld16B(ap0); a0.q[1] = ld16B(ap0 + 16);
        a1.q[0] = ld16B(ap1); a1.q[1] = ld16B(ap1 + 16);
        const _Float16* wp = wrow + k0 + bK;
        b.q[0] = ld16B(wp); b.q[1] = ld16B(wp + 16);
        acc0 = __builtin_amdgcn_wmma_f32_16x16x32_f16(false, a0.h, false, b.h,
                                                      (short)0, acc0, false, false);
        acc1 = __builtin_amdgcn_wmma_f32_16x16x32_f16(false, a1.h, false, b.h,
                                                      (short)0, acc1, false, false);
    }

    const int col = ntile * 16 + (lane & 15);
    const float bias = by[col];
    const int rb0 = mg * 32 + (lane >> 4) * 8;
#pragma unroll
    for (int v = 0; v < 8; ++v) {
        out[(size_t)(rb0 + v) * VOCAB + col]      = acc0[v] + bias;
        out[(size_t)(rb0 + 16 + v) * VOCAB + col] = acc1[v] + bias;
    }
}

// ---------------------------------------------------------------------------
// Host launcher
// ---------------------------------------------------------------------------
extern "C" void kernel_launch(void* const* d_in, const int* in_sizes, int n_in,
                              void* d_out, int out_size, void* d_ws, size_t ws_size,
                              hipStream_t stream) {
    (void)in_sizes; (void)n_in; (void)out_size; (void)ws_size;

    // ---- inputs (setup_inputs() order)
    const int*   texts = (const int*)  d_in[0];
    const float* emb   = (const float*)d_in[1];
    const float* Wf0 = (const float*)d_in[2];  const float* bf0 = (const float*)d_in[3];
    const float* Wi0 = (const float*)d_in[4];  const float* bi0 = (const float*)d_in[5];
    const float* Wo0 = (const float*)d_in[6];  const float* bo0 = (const float*)d_in[7];
    const float* Wc0 = (const float*)d_in[8];  const float* bc0 = (const float*)d_in[9];
    const float* Wf1 = (const float*)d_in[10]; const float* bf1 = (const float*)d_in[11];
    const float* Wi1 = (const float*)d_in[12]; const float* bi1 = (const float*)d_in[13];
    const float* Wo1 = (const float*)d_in[14]; const float* bo1 = (const float*)d_in[15];
    const float* Wc1 = (const float*)d_in[16]; const float* bc1 = (const float*)d_in[17];
    const float* Wy  = (const float*)d_in[18]; const float* by  = (const float*)d_in[19];

    // ---- workspace carve-out (256B aligned)
    char* ws = (char*)d_ws;
    size_t off = 0;
    auto carve = [&](size_t bytes) -> void* {
        void* p = ws + off;
        off += (bytes + 255) & ~(size_t)255;
        return p;
    };

    const size_t K0 = HIDDEN + EMBED;    // 1536
    const size_t K1 = 2 * HIDDEN;        // 2048

    _Float16* xs  = (_Float16*)carve((size_t)SEQLEN * BATCH * EMBED * 2);
    _Float16* w0h[4], *w1h[4];
    for (int g = 0; g < 4; ++g) w0h[g] = (_Float16*)carve((size_t)HIDDEN * K0 * 2);
    for (int g = 0; g < 4; ++g) w1h[g] = (_Float16*)carve((size_t)HIDDEN * K1 * 2);
    _Float16* wyh = (_Float16*)carve((size_t)VOCAB * HIDDEN * 2);
    _Float16* h0b[2] = { (_Float16*)carve((size_t)BATCH * HIDDEN * 2),
                         (_Float16*)carve((size_t)BATCH * HIDDEN * 2) };
    _Float16* h1b[2] = { (_Float16*)carve((size_t)BATCH * HIDDEN * 2),
                         (_Float16*)carve((size_t)BATCH * HIDDEN * 2) };
    float* c0 = (float*)carve((size_t)BATCH * HIDDEN * 4);
    float* c1 = (float*)carve((size_t)BATCH * HIDDEN * 4);

    // ---- zero-init recurrent state (every call; deterministic)
    hipMemsetAsync(h0b[0], 0, (size_t)BATCH * HIDDEN * 2, stream);
    hipMemsetAsync(h1b[0], 0, (size_t)BATCH * HIDDEN * 2, stream);
    hipMemsetAsync(c0, 0, (size_t)BATCH * HIDDEN * 4, stream);
    hipMemsetAsync(c1, 0, (size_t)BATCH * HIDDEN * 4, stream);

    // ---- fp32 -> fp16 weight conversions
    auto convert = [&](const float* src, _Float16* dst, size_t n) {
        int blocks = (int)((n + 1023) / 1024);
        if (blocks > 65535) blocks = 65535;
        convert_f32_f16_kernel<<<blocks, 256, 0, stream>>>(src, dst, n);
    };
    convert(Wf0, w0h[0], (size_t)HIDDEN * K0);
    convert(Wi0, w0h[1], (size_t)HIDDEN * K0);
    convert(Wo0, w0h[2], (size_t)HIDDEN * K0);
    convert(Wc0, w0h[3], (size_t)HIDDEN * K0);
    convert(Wf1, w1h[0], (size_t)HIDDEN * K1);
    convert(Wi1, w1h[1], (size_t)HIDDEN * K1);
    convert(Wo1, w1h[2], (size_t)HIDDEN * K1);
    convert(Wc1, w1h[3], (size_t)HIDDEN * K1);
    convert(Wy,  wyh,    (size_t)VOCAB * HIDDEN);

    // ---- embedding gather + convert
    embed_kernel<<<SEQLEN * BATCH, 128, 0, stream>>>(texts, emb, xs);

    // ---- recurrent loop: interleave layer0/layer1 per timestep
    const dim3 grid(BATCH / 32, HIDDEN / 16);  // (2, 64)
    for (int t = 0; t < SEQLEN; ++t) {
        const _Float16* x_t = xs + (size_t)t * BATCH * EMBED;
        _Float16* h0in  = h0b[t & 1];
        _Float16* h0out = h0b[(t + 1) & 1];
        lstm_step_kernel<<<grid, 32, 0, stream>>>(
            x_t, EMBED, h0in, c0,
            w0h[0], w0h[1], w0h[2], w0h[3],
            bf0, bi0, bo0, bc0,
            h0out, c0);

        _Float16* h1in  = h1b[t & 1];
        _Float16* h1out = h1b[(t + 1) & 1];
        lstm_step_kernel<<<grid, 32, 0, stream>>>(
            h0out, HIDDEN, h1in, c1,
            w1h[0], w1h[1], w1h[2], w1h[3],
            bf1, bi1, bo1, bc1,
            h1out, c1);
    }

    // ---- final projection from h1 at t = SEQLEN-1 (lives in h1b[SEQLEN & 1])
    proj_kernel<<<dim3(BATCH / 32, VOCAB / 16), 32, 0, stream>>>(
        h1b[SEQLEN & 1], wyh, by, (float*)d_out);
}
